// SSA_28647431864329
// MI455X (gfx1250) — compile-verified, matrix operand order checked by reading, not run on hardware
//
#include <hip/hip_runtime.h>

typedef __attribute__((ext_vector_type(16))) _Float16 v16h;
typedef __attribute__((ext_vector_type(8)))  _Float16 v8h;
typedef __attribute__((ext_vector_type(8)))  float    v8f;
typedef __attribute__((ext_vector_type(4)))  int      v4i;

#define NHEADS 12
#define DHEAD  64
#define CH     768
#define C3     2304
#define SP     2744          // 14*14*14 tokens per batch
#define BATCH  2
#define MTOT   (BATCH*SP)    // 5488 rows
#define MSUP   ((MTOT + 127) / 128)   // 43 supertiles of 128 rows

// ---------------------------------------------------------------------------
// Async global->LDS copy (CDNA5 GLOBAL_LOAD_ASYNC_TO_LDS_B128, ASYNCcnt).
// Probe result: builtin exists, signature is
//   (v4i addrspace(1)* src, v4i addrspace(3)* dst, imm offset, imm cpol)
// Guarded by __has_builtin with a synchronous fallback so compile never breaks.
// ---------------------------------------------------------------------------
#if defined(__has_builtin)
#if __has_builtin(__builtin_amdgcn_global_load_async_to_lds_b128)
#define HAS_ASYNC_LDS 1
#endif
#endif
#ifndef HAS_ASYNC_LDS
#define HAS_ASYNC_LDS 0
#endif

typedef __attribute__((address_space(1))) v4i GV4;
typedef __attribute__((address_space(3))) v4i LV4;

__device__ __forceinline__ void async_copy_b128(const _Float16* gp, _Float16* lp) {
#if HAS_ASYNC_LDS
  __builtin_amdgcn_global_load_async_to_lds_b128((GV4*)gp, (LV4*)lp, 0, 0);
#else
  *reinterpret_cast<v8h*>(lp) = *reinterpret_cast<const v8h*>(gp);
#endif
}

__device__ __forceinline__ void async_wait_le2() {
#if HAS_ASYNC_LDS
  __asm__ volatile("s_wait_asynccnt 2" ::: "memory");
#else
  __asm__ volatile("" ::: "memory");
#endif
}

__device__ __forceinline__ void async_wait_le0() {
#if HAS_ASYNC_LDS
  __asm__ volatile("s_wait_asynccnt 0" ::: "memory");
#else
  __asm__ volatile("" ::: "memory");
#endif
}

// ---------------------------------------------------------------------------
// Fragment loaders. Per CDNA5 ISA 7.12.2, for V_WMMA_F32_16X16X32_F16 both the
// A (16x32, lane = row) and B (32x16, lane = column) fragments hold, per lane,
// two contiguous 8xf16 runs at k = 8*(lane>>4) and k = 16 + 8*(lane>>4).
// So a fragment is exactly two 16-byte loads from a row-major f16 array.
// ---------------------------------------------------------------------------
__device__ __forceinline__ v16h frag_from_ptr(const _Float16* p, int hf) {
  v8h lo = *reinterpret_cast<const v8h*>(p + hf * 8);
  v8h hi = *reinterpret_cast<const v8h*>(p + 16 + hf * 8);
  return __builtin_shufflevector(lo, hi, 0,1,2,3,4,5,6,7,8,9,10,11,12,13,14,15);
}

__device__ __forceinline__ v16h load_frag16(const _Float16* tile, int stride) {
  int l = threadIdx.x & 31;
  return frag_from_ptr(tile + (size_t)(l & 15) * stride, l >> 4);
}

__device__ __forceinline__ v8f wmma_f16(v16h a, v16h b, v8f c) {
  return __builtin_amdgcn_wmma_f32_16x16x32_f16(false, a, false, b,
                                                (short)0, c, false, false);
}

// ---------------------------------------------------------------------------
// Weight conversion fp32 -> fp16
// ---------------------------------------------------------------------------
__global__ void f32_to_f16(const float* __restrict__ src,
                           _Float16* __restrict__ dst, int n) {
  int i = blockIdx.x * 256 + threadIdx.x;
  if (i < n) dst[i] = (_Float16)src[i];
}

// ---------------------------------------------------------------------------
// LayerNorm + (b c d h w) -> (b n c) transpose, output f16 token-major rows
// ---------------------------------------------------------------------------
__global__ void __launch_bounds__(256) ln_kernel(const float* __restrict__ x,
                                                 const float* __restrict__ nw,
                                                 const float* __restrict__ nb,
                                                 _Float16* __restrict__ tn) {
  __shared__ float s1[256], s2[256];
  int tok = blockIdx.x;
  int b = tok / SP, sp = tok % SP;
  int t = threadIdx.x;
  const float* xb = x + (size_t)b * CH * SP + sp;
  float v0 = xb[(size_t)(t      ) * SP];
  float v1 = xb[(size_t)(t + 256) * SP];
  float v2 = xb[(size_t)(t + 512) * SP];
  s1[t] = v0 + v1 + v2;
  s2[t] = v0 * v0 + v1 * v1 + v2 * v2;
  __syncthreads();
  for (int o = 128; o > 0; o >>= 1) {
    if (t < o) { s1[t] += s1[t + o]; s2[t] += s2[t + o]; }
    __syncthreads();
  }
  float mean = s1[0] * (1.0f / CH);
  float var  = s2[0] * (1.0f / CH) - mean * mean;
  float rstd = rsqrtf(var + 1e-5f);
  _Float16* o = tn + (size_t)tok * CH;
  o[t]       = (_Float16)((v0 - mean) * rstd * nw[t]       + nb[t]);
  o[t + 256] = (_Float16)((v1 - mean) * rstd * nw[t + 256] + nb[t + 256]);
  o[t + 512] = (_Float16)((v2 - mean) * rstd * nw[t + 512] + nb[t + 512]);
}

// ---------------------------------------------------------------------------
// Stage a 64-col x 32-k weight tile (4 KB) into LDS: 256 x 16B segments,
// 2 async b128 copies per thread (128 threads).
// ---------------------------------------------------------------------------
__device__ __forceinline__ void stage_B(const _Float16* __restrict__ w,
                                        int n0, int kc, _Float16* dst) {
  int t = threadIdx.x;
#pragma unroll
  for (int j = 0; j < 2; j++) {
    int lin = j * 128 + t;       // 0..255
    int row = lin >> 2;          // output column 0..63
    int seg = lin & 3;           // 16B segment within 32 k-values
    async_copy_b128(w + (size_t)(n0 + row) * CH + kc + seg * 8,
                    dst + row * 32 + seg * 8);
  }
}

// ---------------------------------------------------------------------------
// Generic epilogue-less GEMM core: 128-thread block computes 128 rows x 64
// cols; each wave owns 32 rows. B tile double-buffered in LDS via async DMA.
// acc[0][i] rows m0..m0+15, acc[1][i] rows m0+16..m0+31 (i = 16-col subtile).
// ---------------------------------------------------------------------------
__device__ __forceinline__ void gemm_core(const _Float16* __restrict__ a_mat,
                                          const _Float16* __restrict__ w,
                                          int m_wave, int n0,
                                          _Float16* Bt0, _Float16* Bt1,
                                          v8f acc[2][4]) {
  int l = threadIdx.x & 31, ln15 = l & 15, hf = l >> 4;
  int ra0 = m_wave + ln15;      if (ra0 >= MTOT) ra0 = MTOT - 1;
  int ra1 = m_wave + 16 + ln15; if (ra1 >= MTOT) ra1 = MTOT - 1;
  const _Float16* ap0 = a_mat + (size_t)ra0 * CH;
  const _Float16* ap1 = a_mat + (size_t)ra1 * CH;

  stage_B(w, n0, 0, Bt0);
  for (int kc = 0; kc < CH; kc += 32) {
    _Float16* cur = ((kc >> 5) & 1) ? Bt1 : Bt0;
    _Float16* nxt = ((kc >> 5) & 1) ? Bt0 : Bt1;
    if (kc + 32 < CH) {
      stage_B(w, n0, kc + 32, nxt);
      async_wait_le2();          // previous chunk's 2 copies complete
    } else {
      async_wait_le0();
    }
    __syncthreads();             // staged tile visible to all waves

    v16h a0 = frag_from_ptr(ap0 + kc, hf);
    v16h a1 = frag_from_ptr(ap1 + kc, hf);
#pragma unroll
    for (int i = 0; i < 4; i++) {
      v16h b = load_frag16(cur + i * 16 * 32, 32);   // ds_load_b128 pairs
      acc[0][i] = wmma_f16(a0, b, acc[0][i]);
      acc[1][i] = wmma_f16(a1, b, acc[1][i]);
    }
    __syncthreads();             // all reads done before buffer reuse
  }
}

// ---------------------------------------------------------------------------
// QKV projection. M=5488, N=2304, K=768. Writes Q,K as [bh][token][64] and
// V transposed as [bh][64][token].
// ---------------------------------------------------------------------------
__global__ void __launch_bounds__(128) qkv_gemm(const _Float16* __restrict__ tn,
                                                const _Float16* __restrict__ w,
                                                _Float16* __restrict__ q,
                                                _Float16* __restrict__ k,
                                                _Float16* __restrict__ vt) {
  __shared__ __align__(16) _Float16 Bt[2][64 * 32];
  int wave = threadIdx.x >> 5;
  int l = threadIdx.x & 31, ln15 = l & 15, hf = l >> 4;
  int m_wave = blockIdx.x * 128 + wave * 32;
  int n0 = blockIdx.y * 64;

  v8f z = {0.f,0.f,0.f,0.f,0.f,0.f,0.f,0.f};
  v8f acc[2][4] = {{z, z, z, z}, {z, z, z, z}};

  gemm_core(tn, w, m_wave, n0, &Bt[0][0], &Bt[1][0], acc);

#pragma unroll
  for (int s = 0; s < 2; s++) {
#pragma unroll
    for (int i = 0; i < 4; i++) {
      int n = n0 + i * 16 + ln15;
#pragma unroll
      for (int r = 0; r < 8; r++) {
        int m = m_wave + s * 16 + r + 8 * hf;
        if (m < MTOT) {
          int b = m / SP, tok = m % SP;
          _Float16 val = (_Float16)acc[s][i][r];
          if (n < CH) {
            int head = n >> 6, d = n & 63;
            q[(((size_t)(b * NHEADS + head)) * SP + tok) * DHEAD + d] = val;
          } else if (n < 2 * CH) {
            int nn = n - CH, head = nn >> 6, d = nn & 63;
            k[(((size_t)(b * NHEADS + head)) * SP + tok) * DHEAD + d] = val;
          } else {
            int nn = n - 2 * CH, head = nn >> 6, d = nn & 63;
            vt[(((size_t)(b * NHEADS + head)) * DHEAD + d) * SP + tok] = val;
          }
        }
      }
    }
  }
}

// ---------------------------------------------------------------------------
// Flash attention. One wave per (batch*head, 16-query-row tile). Streams keys
// in tiles of 32, online softmax, O accumulated in 4 f32 C-fragments.
// P (16x32 f16) round-trips through LDS to convert C-layout -> A-layout.
// ---------------------------------------------------------------------------
__global__ void __launch_bounds__(32) attn_kernel(const _Float16* __restrict__ q,
                                                  const _Float16* __restrict__ kk,
                                                  const _Float16* __restrict__ vt,
                                                  _Float16* __restrict__ oh) {
  __shared__ __align__(16) _Float16 Pl[16 * 32];
  int bh = blockIdx.x;                 // 0..23
  int qt = blockIdx.y;                 // 0..171
  int l = threadIdx.x & 31, ln15 = l & 15, hf = l >> 4;
  int m0 = qt * 16;

  const _Float16* qbase = q  + (size_t)bh * SP * DHEAD;
  const _Float16* kbase = kk + (size_t)bh * SP * DHEAD;
  const _Float16* vbase = vt + (size_t)bh * DHEAD * SP;

  int qrow = m0 + ln15; if (qrow >= SP) qrow = SP - 1;
  const _Float16* qp = qbase + (size_t)qrow * DHEAD;
  v16h q0 = frag_from_ptr(qp, hf);
  v16h q1 = frag_from_ptr(qp + 32, hf);

  v8f z = {0.f,0.f,0.f,0.f,0.f,0.f,0.f,0.f};
  v8f oacc[4] = {z, z, z, z};
  float mrow[8], lrow[8];
#pragma unroll
  for (int r = 0; r < 8; r++) { mrow[r] = -1e30f; lrow[r] = 0.f; }

  const float scale = 0.125f;          // 1/sqrt(64)

  for (int ky0 = 0; ky0 < SP; ky0 += 32) {
    v8f s0 = z, s1 = z;
    {
      int kr0 = ky0 + ln15;      if (kr0 >= SP) kr0 = SP - 1;
      const _Float16* kp0 = kbase + (size_t)kr0 * DHEAD;
      s0 = wmma_f16(q0, frag_from_ptr(kp0, hf), s0);
      s0 = wmma_f16(q1, frag_from_ptr(kp0 + 32, hf), s0);
      int kr1 = ky0 + 16 + ln15; if (kr1 >= SP) kr1 = SP - 1;
      const _Float16* kp1 = kbase + (size_t)kr1 * DHEAD;
      s1 = wmma_f16(q0, frag_from_ptr(kp1, hf), s1);
      s1 = wmma_f16(q1, frag_from_ptr(kp1 + 32, hf), s1);
    }
#pragma unroll
    for (int r = 0; r < 8; r++) { s0[r] *= scale; s1[r] *= scale; }
    if (ky0 + ln15 >= SP) {
#pragma unroll
      for (int r = 0; r < 8; r++) s0[r] = -1e30f;
    }
    if (ky0 + 16 + ln15 >= SP) {
#pragma unroll
      for (int r = 0; r < 8; r++) s1[r] = -1e30f;
    }

#pragma unroll
    for (int r = 0; r < 8; r++) {
      float v = fmaxf(s0[r], s1[r]);
      v = fmaxf(v, __shfl_xor(v, 1));
      v = fmaxf(v, __shfl_xor(v, 2));
      v = fmaxf(v, __shfl_xor(v, 4));
      v = fmaxf(v, __shfl_xor(v, 8));
      float mnew  = fmaxf(mrow[r], v);
      float alpha = __expf(mrow[r] - mnew);
      float e0 = __expf(s0[r] - mnew);
      float e1 = __expf(s1[r] - mnew);
      int row = r + 8 * hf;
      Pl[row * 32 + ln15]      = (_Float16)e0;
      Pl[row * 32 + 16 + ln15] = (_Float16)e1;
      float rs = e0 + e1;
      rs += __shfl_xor(rs, 1);
      rs += __shfl_xor(rs, 2);
      rs += __shfl_xor(rs, 4);
      rs += __shfl_xor(rs, 8);
      lrow[r] = lrow[r] * alpha + rs;
      mrow[r] = mnew;
#pragma unroll
      for (int dc = 0; dc < 4; dc++) oacc[dc][r] *= alpha;
    }

    __asm__ volatile("s_wait_dscnt 0" ::: "memory");

    v16h pfrag = frag_from_ptr(&Pl[(size_t)ln15 * 32], hf);

#pragma unroll
    for (int dc = 0; dc < 4; dc++) {
      const _Float16* vrow = vbase + (size_t)(dc * 16 + ln15) * SP + ky0;
      oacc[dc] = wmma_f16(pfrag, frag_from_ptr(vrow, hf), oacc[dc]);
    }
  }

  int b = bh / NHEADS, head = bh % NHEADS;
#pragma unroll
  for (int dc = 0; dc < 4; dc++) {
    int col = head * DHEAD + dc * 16 + ln15;
#pragma unroll
    for (int r = 0; r < 8; r++) {
      int m = m0 + r + 8 * hf;
      if (m < SP) {
        float val = oacc[dc][r] / lrow[r];
        oh[((size_t)(b * SP + m)) * CH + col] = (_Float16)val;
      }
    }
  }
}

// ---------------------------------------------------------------------------
// Output projection + bias + residual, scattered back to (b c d h w) fp32.
// ---------------------------------------------------------------------------
__global__ void __launch_bounds__(128) out_gemm(const _Float16* __restrict__ oh,
                                                const _Float16* __restrict__ w,
                                                const float* __restrict__ bias,
                                                const float* __restrict__ x,
                                                float* __restrict__ out) {
  __shared__ __align__(16) _Float16 Bt[2][64 * 32];
  int wave = threadIdx.x >> 5;
  int l = threadIdx.x & 31, ln15 = l & 15, hf = l >> 4;
  int m_wave = blockIdx.x * 128 + wave * 32;
  int n0 = blockIdx.y * 64;

  v8f z = {0.f,0.f,0.f,0.f,0.f,0.f,0.f,0.f};
  v8f acc[2][4] = {{z, z, z, z}, {z, z, z, z}};

  gemm_core(oh, w, m_wave, n0, &Bt[0][0], &Bt[1][0], acc);

#pragma unroll
  for (int s = 0; s < 2; s++) {
#pragma unroll
    for (int i = 0; i < 4; i++) {
      int n = n0 + i * 16 + ln15;
      float bi = bias[n];
#pragma unroll
      for (int r = 0; r < 8; r++) {
        int m = m_wave + s * 16 + r + 8 * hf;
        if (m < MTOT) {
          int b = m / SP, tok = m % SP;
          size_t idx = ((size_t)(b * CH + n)) * SP + tok;
          out[idx] = acc[s][i][r] + bi + x[idx];
        }
      }
    }
  }
}

// ---------------------------------------------------------------------------
extern "C" void kernel_launch(void* const* d_in, const int* in_sizes, int n_in,
                              void* d_out, int out_size, void* d_ws, size_t ws_size,
                              hipStream_t stream) {
  const float* x      = (const float*)d_in[0];
  const float* norm_w = (const float*)d_in[1];
  const float* norm_b = (const float*)d_in[2];
  const float* qkv_w  = (const float*)d_in[3];
  const float* out_w  = (const float*)d_in[4];
  const float* out_b  = (const float*)d_in[5];
  float* out = (float*)d_out;

  char* ws = (char*)d_ws;
  size_t off = 0;
  auto take = [&](size_t bytes) {
    char* p = ws + off;
    off += (bytes + 255) & ~(size_t)255;
    return p;
  };
  _Float16* tn   = (_Float16*)take((size_t)MTOT * CH * 2);
  _Float16* qw_h = (_Float16*)take((size_t)C3 * CH * 2);
  _Float16* ow_h = (_Float16*)take((size_t)CH * CH * 2);
  _Float16* qarr = (_Float16*)take((size_t)BATCH * NHEADS * SP * DHEAD * 2);
  _Float16* karr = (_Float16*)take((size_t)BATCH * NHEADS * SP * DHEAD * 2);
  _Float16* vtar = (_Float16*)take((size_t)BATCH * NHEADS * DHEAD * SP * 2);
  _Float16* ohm  = (_Float16*)take((size_t)MTOT * CH * 2);

  f32_to_f16<<<(C3 * CH + 255) / 256, 256, 0, stream>>>(qkv_w, qw_h, C3 * CH);
  f32_to_f16<<<(CH * CH + 255) / 256, 256, 0, stream>>>(out_w, ow_h, CH * CH);
  ln_kernel<<<MTOT, 256, 0, stream>>>(x, norm_w, norm_b, tn);
  qkv_gemm<<<dim3(MSUP, C3 / 64), 128, 0, stream>>>(tn, qw_h, qarr, karr, vtar);
  attn_kernel<<<dim3(BATCH * NHEADS, (SP + 15) / 16), 32, 0, stream>>>(qarr, karr, vtar, ohm);
  out_gemm<<<dim3(MSUP, CH / 64), 128, 0, stream>>>(ohm, ow_h, out_b, x, out);
}
